// MLPPredictor_25280177504629
// MI455X (gfx1250) — matrix operand order repaired and checked by reading, not compiled, not run on hardware
//
#include <hip/hip_runtime.h>
#include <hip/hip_bf16.h>

// ---------------------------------------------------------------------------
// Edge MLP:  score[e] = W3 @ relu(W2 @ relu(W1 @ concat(x[src],x[dst]) + b1) + b2) + b3
// bf16 WMMA (f32 accumulate) for the two 256-wide layers.
// ---------------------------------------------------------------------------

typedef __attribute__((ext_vector_type(16))) __bf16 v16bf;
typedef __attribute__((ext_vector_type(8)))  __bf16 v8bf;
typedef __attribute__((ext_vector_type(8)))  float  v8f;

union Frag16 { v16bf v; v8bf h[2]; };

#define D_IN  256
#define D_HID 256
#define MT    32        // edges per workgroup
#define A1_STRIDE (512 + 8)   // bf16 elems; 1040B row stride -> 4-bank shift, conflict free
#define A2_STRIDE (256 + 8)   // 528B row stride -> 4-bank shift

// ---------------------------------------------------------------------------
// Pre-swizzle W ([256, K] row-major f32) into per-lane B-fragment order, bf16.
// Fragment (ks, nt): 32 lanes x 16 contiguous bf16 per lane.
//   lane = c*16 + nlo  holds  B[k = ks*32 + c*16 + j][n = nt*16 + nlo], j=0..15
//   (B[k][n] == W[n][k] since the layer computes h @ W^T)
// ---------------------------------------------------------------------------
__global__ void swizzle_w_kernel(const float* __restrict__ W,
                                 __bf16* __restrict__ out,
                                 int K, int total) {
    int o = blockIdx.x * 256 + threadIdx.x;
    if (o >= total) return;
    int j    = o & 15;
    int lane = (o >> 4) & 31;
    int frag = o >> 9;
    int ks   = frag >> 4;      // K-step (32 K per step)
    int nt   = frag & 15;      // N tile (16 cols per tile)
    int c    = lane >> 4;
    int n    = nt * 16 + (lane & 15);
    int k    = ks * 32 + c * 16 + j;
    out[o] = (__bf16)W[n * K + k];
}

// ---------------------------------------------------------------------------
// One WMMA layer: D = relu(A @ B + bias), A: [32, 32*KSTEPS] bf16 in LDS,
// B fragments pre-swizzled in global. Each wave covers 32 output columns.
// ---------------------------------------------------------------------------
template <int KSTEPS>
__device__ __forceinline__ void mlp_layer(const __bf16* __restrict__ Alds, int aStride,
                                          const __bf16* __restrict__ Ws,
                                          const float* __restrict__ bias,
                                          __bf16* __restrict__ Dlds, int dStride,
                                          int wave, int lane) {
    const int c   = lane >> 4;     // K-half selector for A/B frags, M-half for C
    const int nlo = lane & 15;     // row (A) / col (B,C) within tile
    const int nt0 = wave * 2;      // first of this wave's two N tiles

    v8f acc[2][2] = {};            // [mSub][nSub]

#pragma unroll
    for (int ks = 0; ks < KSTEPS; ++ks) {
        const int k0 = ks * 32;
        Frag16 a[2], b[2];
#pragma unroll
        for (int ms = 0; ms < 2; ++ms) {
            const __bf16* ap = Alds + (ms * 16 + nlo) * aStride + k0 + c * 16;
            a[ms].h[0] = *(const v8bf*)(ap);        // K = k0 + c*16   .. +7
            a[ms].h[1] = *(const v8bf*)(ap + 8);    // K = k0+c*16+8 .. +7
        }
        const __bf16* bp = Ws + ((size_t)(ks * 16 + nt0) * 32 + lane) * 16;
#pragma unroll
        for (int ns = 0; ns < 2; ++ns)
            b[ns].v = *(const v16bf*)(bp + ns * 512);   // next N tile = +512 elems
#pragma unroll
        for (int ms = 0; ms < 2; ++ms)
#pragma unroll
            for (int ns = 0; ns < 2; ++ns)
                acc[ms][ns] = __builtin_amdgcn_wmma_f32_16x16x32_bf16(
                    false, a[ms].v, false, b[ns].v, 0, acc[ms][ns], false, false);
    }

    // epilogue: bias + relu, repack to bf16 A-tile for next layer
#pragma unroll
    for (int ms = 0; ms < 2; ++ms) {
#pragma unroll
        for (int ns = 0; ns < 2; ++ns) {
            const int n  = (nt0 + ns) * 16 + nlo;
            const float bn = bias[n];
#pragma unroll
            for (int r = 0; r < 8; ++r) {
                float v = acc[ms][ns][r] + bn;
                v = fmaxf(v, 0.0f);
                Dlds[(ms * 16 + c * 8 + r) * dStride + n] = (__bf16)v;
            }
        }
    }
}

// ---------------------------------------------------------------------------
// Main kernel: one workgroup = 32 edges.
// ---------------------------------------------------------------------------
__global__ __launch_bounds__(256) void edge_mlp_kernel(
    const float* __restrict__ x,
    const long long* __restrict__ src,
    const long long* __restrict__ dst,
    const float* __restrict__ b1,
    const float* __restrict__ b2,
    const float* __restrict__ W3,
    const float* __restrict__ b3,
    const __bf16* __restrict__ W1s,
    const __bf16* __restrict__ W2s,
    float* __restrict__ out,
    int nEdges) {
    __shared__ __bf16 A1[MT][A1_STRIDE];
    __shared__ __bf16 A2[MT][A2_STRIDE];
    __shared__ __bf16 A3[MT][A2_STRIDE];

    const int tid  = threadIdx.x;
    const int wave = tid >> 5;
    const int lane = tid & 31;
    const int eBase = blockIdx.x * MT;

    // ---- stage concat(x[src], x[dst]) -> bf16 A-tile in LDS ----
    {
        const int rid  = tid >> 2;      // 0..63: (edge, half)
        const int q    = tid & 3;       // quarter of a 256-col half
        const int e    = rid >> 1;
        const int half = rid & 1;
        const int eg   = eBase + e;
        long long node = 0;
        if (eg < nEdges) node = half ? dst[eg] : src[eg];
        const float4* xr = (const float4*)(x + (size_t)node * D_IN + q * 64);
#pragma unroll
        for (int i = 0; i < 16; ++i) {
            float4 v = xr[i];
            const int cidx = half * 256 + q * 64 + i * 4;
            A1[e][cidx + 0] = (__bf16)v.x;
            A1[e][cidx + 1] = (__bf16)v.y;
            A1[e][cidx + 2] = (__bf16)v.z;
            A1[e][cidx + 3] = (__bf16)v.w;
        }
    }
    __syncthreads();

    // ---- layer 1: [32,512] @ [512,256] ----
    mlp_layer<16>(&A1[0][0], A1_STRIDE, W1s, b1, &A2[0][0], A2_STRIDE, wave, lane);
    __syncthreads();

    // ---- layer 2: [32,256] @ [256,256] ----
    mlp_layer<8>(&A2[0][0], A2_STRIDE, W2s, b2, &A3[0][0], A2_STRIDE, wave, lane);
    __syncthreads();

    // ---- layer 3: [32,256] @ [256,1] dot + shuffle reduce ----
    {
        const int e  = wave * 4 + (lane >> 3);   // 8 waves * 4 edges = 32
        const int c0 = (lane & 7) * 32;          // 8 lanes per edge, 32 cols each
        float s = 0.0f;
#pragma unroll 8
        for (int j = 0; j < 32; ++j)
            s += (float)A3[e][c0 + j] * W3[c0 + j];
        s += __shfl_xor(s, 1, 32);
        s += __shfl_xor(s, 2, 32);
        s += __shfl_xor(s, 4, 32);
        if ((lane & 7) == 0 && (eBase + e) < nEdges)
            out[eBase + e] = s + b3[0];
    }
}

// ---------------------------------------------------------------------------
// Launch
// ---------------------------------------------------------------------------
extern "C" void kernel_launch(void* const* d_in, const int* in_sizes, int n_in,
                              void* d_out, int out_size, void* d_ws, size_t ws_size,
                              hipStream_t stream) {
    const float*      x   = (const float*)d_in[0];
    const long long*  src = (const long long*)d_in[1];
    const long long*  dst = (const long long*)d_in[2];
    const float*      W1  = (const float*)d_in[3];   // [256, 512]
    const float*      b1  = (const float*)d_in[4];
    const float*      W2  = (const float*)d_in[5];   // [256, 256]
    const float*      b2  = (const float*)d_in[6];
    const float*      W3  = (const float*)d_in[7];   // [1, 256]
    const float*      b3  = (const float*)d_in[8];
    float*            out = (float*)d_out;

    const int nEdges = in_sizes[1];

    // workspace: swizzled bf16 weights
    __bf16* W1s = (__bf16*)d_ws;                 // 256*512 = 131072 elems (256 KB)
    __bf16* W2s = W1s + 256 * 512;               // 256*256 =  65536 elems (128 KB)

    swizzle_w_kernel<<<(256 * 512) / 256, 256, 0, stream>>>(W1, W1s, 512, 256 * 512);
    swizzle_w_kernel<<<(256 * 256) / 256, 256, 0, stream>>>(W2, W2s, 256, 256 * 256);

    const int tiles = (nEdges + MT - 1) / MT;
    edge_mlp_kernel<<<tiles, 256, 0, stream>>>(x, src, dst, b1, b2, W3, b3,
                                               W1s, W2s, out, nEdges);
}
